// WindowAttention_22144851378685
// MI455X (gfx1250) — matrix operand order repaired and checked by reading, not compiled, not run on hardware
//
#include <hip/hip_runtime.h>
#include <hip/hip_bf16.h>

// MI455X / gfx1250 fused axial-attention kernel (wave32, WMMA bf16).
//
// All GEMMs are oriented so the C/D tile comes out "transposed": the lane
// index maps to the row we store, and the 8 accumulator VGPRs map to
// consecutive columns -> every C-tile store is a single packed 16B store.
//
// Fragment layouts used here (wave32; C/D per CDNA5 ISA 7.12.2):
//  A-frag (16x32 bf16): lane m = lane&15; 16 consecutive halfs at
//                       K=kb+(hi?16:0).
//  B-frag (32x16 bf16): lane n = lane&15; 16 consecutive halfs at
//                       K=kb+(hi?16:0).
//  C/D (16x16 f32):     lane n = lane&15 -> N; vgpr r -> M = r + 8*hi.

typedef __bf16 bf16_t;
typedef __attribute__((ext_vector_type(16))) __bf16 v16bf;
typedef __attribute__((ext_vector_type(8)))  __bf16 v8bf;
typedef __attribute__((ext_vector_type(4)))  __bf16 v4bf;
typedef __attribute__((ext_vector_type(8)))  float  v8f;

#define BATCH 4
#define HDIM 256
#define WDIM 256
#define CDIM 128
#define C3   (3 * CDIM)

// LDS strides in bf16 elements (+8 pad: conflict-free, keeps 16B alignment)
#define LDK 136   // for LX (x col) and LK (k), 128 cols
#define LDG 264   // for LV (v^T) and per-wave scratch, 256 cols

// LDS region byte offsets
#define OFF_LX   0
#define SZ_LX    (HDIM * LDK * 2)            // 69632
#define OFF_LK   (OFF_LX + SZ_LX)
#define SZ_LK    (HDIM * LDK * 2)            // 69632
#define OFF_LV   (OFF_LK + SZ_LK)
#define SZ_LV    (CDIM * LDG * 2)            // 67584
#define OFF_SCR  (OFF_LV + SZ_LV)
#define SCR_WAVE (16 * LDG * 2)              // 8448 per-wave q/P/O scratch
#define SMEM_BYTES (OFF_SCR + 8 * SCR_WAVE)  // 274432 B < 320 KB

// Workspace layout (bytes): pre-transposed bf16 weights, Wt[col][k]
#define WS_WTQ 0                              // bf16 [C3][CDIM]   = 98304 B
#define WS_WTP (C3 * CDIM * 2)                // bf16 [CDIM][CDIM] = 32768 B

static __device__ inline v16bf catbf(v8bf lo, v8bf hi) {
  return __builtin_shufflevector(lo, hi, 0, 1, 2, 3, 4, 5, 6, 7,
                                 8, 9, 10, 11, 12, 13, 14, 15);
}

// A-fragment from a row-major LDS matrix; `row` = start of this lane's M row.
static __device__ inline v16bf ldsA(const bf16_t* row, int hi16, int kb) {
  const bf16_t* p = row + kb + hi16 * 16;
  v8bf lo = *(const v8bf*)(p);
  v8bf hi = *(const v8bf*)(p + 8);
  return catbf(lo, hi);
}

// B-fragment: `row` = start of this lane's N column data (K contiguous).
static __device__ inline v16bf ldsB(const bf16_t* row, int hi16, int kb) {
  const bf16_t* p = row + kb + hi16 * 16;
  v8bf lo = *(const v8bf*)(p);
  v8bf hi = *(const v8bf*)(p + 8);
  return catbf(lo, hi);
}

// A-fragment from pre-transposed bf16 weights Wt[col][k] (k contiguous):
// two 16B global loads per lane; `row` selects the output channel.
static __device__ inline v16bf ldwA(const bf16_t* __restrict__ Wt, int row,
                                    int hi16, int kb) {
  const bf16_t* p = Wt + (size_t)row * CDIM + kb + hi16 * 16;
  v8bf lo = *(const v8bf*)(p);
  v8bf hi = *(const v8bf*)(p + 8);
  return catbf(lo, hi);
}

// B-fragment from Wt[col][k]: lane's fixed column, 16 consecutive k.
static __device__ inline v16bf ldwB(const bf16_t* __restrict__ Wt, int col,
                                    int k0) {
  const bf16_t* p = Wt + (size_t)col * CDIM + k0;
  v8bf lo = *(const v8bf*)(p);
  v8bf hi = *(const v8bf*)(p + 8);
  return catbf(lo, hi);
}

static __device__ inline v8f wmma_bf16(v16bf a, v16bf b, v8f c) {
  return __builtin_amdgcn_wmma_f32_16x16x32_bf16(
      /*neg_a=*/false, a, /*neg_b=*/false, b,
      /*c_mod=*/(short)0, c, /*reuse_a=*/false, /*reuse_b=*/false);
}

// Add 8 consecutive bias values (this lane's packed output channels).
static __device__ inline void addbias8(v8f& o, const float* __restrict__ b,
                                       int base) {
  const float4 b0 = *(const float4*)(b + base);
  const float4 b1 = *(const float4*)(b + base + 4);
  o[0] += b0.x; o[1] += b0.y; o[2] += b0.z; o[3] += b0.w;
  o[4] += b1.x; o[5] += b1.y; o[6] += b1.z; o[7] += b1.w;
}

static __device__ inline v8bf cvt8(v8f o) {
  v8bf pk;
#pragma unroll
  for (int r = 0; r < 8; ++r) pk[r] = (__bf16)o[r];
  return pk;
}

// ---- prep: transpose + convert weights to bf16 in workspace --------------
__global__ __launch_bounds__(256)
void wprep_kernel(const float* __restrict__ Wqkv,
                  const float* __restrict__ Wproj,
                  bf16_t* __restrict__ WtQ,   // [C3][CDIM]
                  bf16_t* __restrict__ WtP) { // [CDIM][CDIM]
  int i = blockIdx.x * 256 + threadIdx.x;
  if (i < C3 * CDIM) {
    int col = i >> 7;  // output channel (0..383)
    int k   = i & 127; // input channel
    WtQ[i] = (__bf16)Wqkv[k * C3 + col];
  } else {
    int j = i - C3 * CDIM;
    if (j < CDIM * CDIM) {
      int col = j >> 7;
      int k   = j & 127;
      WtP[j] = (__bf16)Wproj[k * CDIM + col];
    }
  }
}

// ---- main fused kernel ----------------------------------------------------
__global__ __launch_bounds__(256)
void axial_attn_kernel(const float* __restrict__ x,
                       const bf16_t* __restrict__ WtQ,
                       const float* __restrict__ bqkv,
                       const bf16_t* __restrict__ WtP,
                       const float* __restrict__ bproj,
                       float* __restrict__ out) {
  extern __shared__ char smem[];
  bf16_t* LX = (bf16_t*)(smem + OFF_LX);   // x column, bf16 [256][LDK]
  bf16_t* LK = (bf16_t*)(smem + OFF_LK);   // k, row-major [g][c]
  bf16_t* LV = (bf16_t*)(smem + OFF_LV);   // v^T, row-major [c][g]

  const int tid  = threadIdx.x;
  const int lane = tid & 31;
  const int wave = tid >> 5;
  const int l15  = lane & 15;
  const int hi16 = lane >> 4;  // 0 for lanes 0-15, 1 for 16-31

  bf16_t* SCR = (bf16_t*)(smem + OFF_SCR + wave * SCR_WAVE);  // [16][LDG]

  const int bb = blockIdx.x >> 8;   // batch
  const int ww = blockIdx.x & 255;  // column w

  const size_t xrow = (size_t)WDIM * CDIM;  // stride between h rows (floats)
  const float* xcol = x + ((size_t)bb * HDIM * WDIM + ww) * CDIM;

  // Warm L2 with the (globally shared) bf16 weights: 128KB total.
  for (int i = tid; i < (C3 * CDIM) / 128; i += 256)
    __builtin_prefetch(WtQ + i * 128, 0, 2);
  for (int i = tid; i < (CDIM * CDIM) / 128; i += 256)
    __builtin_prefetch(WtP + i * 128, 0, 2);

  // ---- Phase 1: stage x column as bf16 into LX --------------------------
  for (int i = tid; i < HDIM * (CDIM / 4); i += 256) {
    int h  = i >> 5;   // 32 float4 per row
    int c4 = i & 31;
    const float4 v = *(const float4*)(xcol + (size_t)h * xrow + c4 * 4);
    v4bf o;
    o[0] = (__bf16)v.x; o[1] = (__bf16)v.y;
    o[2] = (__bf16)v.z; o[3] = (__bf16)v.w;
    *(v4bf*)&LX[h * LDK + c4 * 4] = o;
  }
  __syncthreads();

  // ---- Phase 2: k and v projections (all waves cooperate) ---------------
  // k^T form: A = Wk^T (global, contiguous), B = x rows (LDS).
  // Result lane -> g row, vgprs -> consecutive c: one packed 16B store.
#pragma unroll 1
  for (int t = 0; t < 16; ++t) {
    int idx = wave * 16 + t;  // 0..127
    int ct  = idx & 7;
    int gt  = idx >> 3;
    v8f acc = {};
#pragma unroll
    for (int kc = 0; kc < 4; ++kc) {
      v16bf a = ldwA(WtQ, CDIM + ct * 16 + l15, hi16, kc * 32);
      v16bf b = ldsB(&LX[(gt * 16 + l15) * LDK], hi16, kc * 32);
      acc = wmma_bf16(a, b, acc);
    }
    addbias8(acc, bqkv, CDIM + ct * 16 + hi16 * 8);
    *(v8bf*)&LK[(gt * 16 + l15) * LDK + ct * 16 + hi16 * 8] = cvt8(acc);
  }
  // v (normal form): A = x rows, B = Wv cols; result lane -> c col,
  // vgprs -> consecutive g: packed store directly into transposed LV[c][g].
#pragma unroll 1
  for (int t = 0; t < 16; ++t) {
    int idx = wave * 16 + t;
    int ct  = idx & 7;
    int gt  = idx >> 3;
    int col = 2 * CDIM + ct * 16 + l15;
    v8f acc = {};
#pragma unroll
    for (int kc = 0; kc < 4; ++kc) {
      v16bf a = ldsA(&LX[(gt * 16 + l15) * LDK], hi16, kc * 32);
      v16bf b = ldwB(WtQ, col, kc * 32 + hi16 * 16);
      acc = wmma_bf16(a, b, acc);
    }
    float bias = bqkv[col];
#pragma unroll
    for (int r = 0; r < 8; ++r) acc[r] += bias;
    *(v8bf*)&LV[(ct * 16 + l15) * LDG + gt * 16 + hi16 * 8] = cvt8(acc);
  }
  __syncthreads();

  // ---- Phase 3: per-wave attention over 2 h-tiles -----------------------
#pragma unroll 1
  for (int hti = 0; hti < 2; ++hti) {
    const int ht = wave + hti * 8;  // h-tile index 0..15

    // q^T form -> SCR row-major [h][c], packed stores.
#pragma unroll 1
    for (int ct = 0; ct < 8; ++ct) {
      v8f acc = {};
#pragma unroll
      for (int kc = 0; kc < 4; ++kc) {
        v16bf a = ldwA(WtQ, ct * 16 + l15, hi16, kc * 32);
        v16bf b = ldsB(&LX[(ht * 16 + l15) * LDK], hi16, kc * 32);
        acc = wmma_bf16(a, b, acc);
      }
      addbias8(acc, bqkv, ct * 16 + hi16 * 8);
      *(v8bf*)&SCR[l15 * LDG + ct * 16 + hi16 * 8] = cvt8(acc);
    }
    // q B-fragments (rows of SCR, reused across all 16 g-tiles)
    v16bf bq[4];
#pragma unroll
    for (int kc = 0; kc < 4; ++kc)
      bq[kc] = ldsB(&SCR[l15 * LDG], hi16, kc * 32);

    // S^T = k @ q^T : lane -> h (= l15), vgprs -> consecutive g.
    v8f acc[16];
#pragma unroll
    for (int gt = 0; gt < 16; ++gt) {
      v8f s = {};
#pragma unroll
      for (int kc = 0; kc < 4; ++kc) {
        v16bf a = ldsA(&LK[(gt * 16 + l15) * LDK], hi16, kc * 32);
        s = wmma_bf16(a, bq[kc], s);
      }
      acc[gt] = s;
    }

    // Softmax over g: purely per-lane (128 values) + one xor16 to merge the
    // partner half-wave lane holding the same h row.
    float m = acc[0][0];
#pragma unroll
    for (int gt = 0; gt < 16; ++gt)
#pragma unroll
      for (int r = 0; r < 8; ++r) m = fmaxf(m, acc[gt][r]);
    m = fmaxf(m, __shfl_xor(m, 16));
    float s = 0.f;
#pragma unroll
    for (int gt = 0; gt < 16; ++gt) {
#pragma unroll
      for (int r = 0; r < 8; ++r) {
        float e = __expf(acc[gt][r] - m);
        acc[gt][r] = e;
        s += e;
      }
    }
    s += __shfl_xor(s, 16);
    const float rinv = s > 0.f ? 1.0f / s : 0.f;  // nan/empty-row mask

    // P (unnormalized, bf16) -> SCR row-major [h][g]: 16 packed stores.
#pragma unroll
    for (int gt = 0; gt < 16; ++gt)
      *(v8bf*)&SCR[l15 * LDG + gt * 16 + hi16 * 8] = cvt8(acc[gt]);

    // O^T = v^T @ P^T : A from LV rows, B from P rows (preloaded).
    v16bf bp[8];
#pragma unroll
    for (int kc = 0; kc < 8; ++kc)
      bp[kc] = ldsB(&SCR[l15 * LDG], hi16, kc * 32);

#pragma unroll 1
    for (int ct = 0; ct < 8; ++ct) {
      v8f o = {};
#pragma unroll
      for (int kc = 0; kc < 8; ++kc) {
        v16bf a = ldsA(&LV[(ct * 16 + l15) * LDG], hi16, kc * 32);
        o = wmma_bf16(a, bp[kc], o);
      }
      // lane -> h, vgprs -> consecutive c: packed store of O[h][c] into SCR
      // (overwrites P; bp already in registers), folding in 1/rowsum.
#pragma unroll
      for (int r = 0; r < 8; ++r) o[r] *= rinv;
      *(v8bf*)&SCR[l15 * LDG + ct * 16 + hi16 * 8] = cvt8(o);
    }

    // out^T = Wproj^T @ O^T : A from WtP (global), B = O rows (preloaded).
    v16bf bo[4];
#pragma unroll
    for (int kc = 0; kc < 4; ++kc)
      bo[kc] = ldsB(&SCR[l15 * LDG], hi16, kc * 32);

    float* outp =
        out + ((size_t)bb * HDIM * WDIM + (size_t)ht * 16 * WDIM + ww) * CDIM;
#pragma unroll 1
    for (int ct = 0; ct < 8; ++ct) {
      v8f o = {};
#pragma unroll
      for (int kc = 0; kc < 4; ++kc) {
        v16bf a = ldwA(WtP, ct * 16 + l15, hi16, kc * 32);
        o = wmma_bf16(a, bo[kc], o);
      }
      addbias8(o, bproj, ct * 16 + hi16 * 8);
      // lane -> h (= l15), vgprs -> 8 consecutive output channels:
      // two contiguous float4 stores per tile.
      float* p = outp + (size_t)l15 * xrow + ct * 16 + hi16 * 8;
      float4 s0; s0.x = o[0]; s0.y = o[1]; s0.z = o[2]; s0.w = o[3];
      float4 s1; s1.x = o[4]; s1.y = o[5]; s1.z = o[6]; s1.w = o[7];
      *(float4*)(p)     = s0;
      *(float4*)(p + 4) = s1;
    }
  }
}

extern "C" void kernel_launch(void* const* d_in, const int* in_sizes, int n_in,
                              void* d_out, int out_size, void* d_ws,
                              size_t ws_size, hipStream_t stream) {
  (void)in_sizes; (void)n_in; (void)out_size; (void)ws_size;
  const float* x     = (const float*)d_in[0];
  const float* Wqkv  = (const float*)d_in[1];
  const float* bqkv  = (const float*)d_in[2];
  const float* Wproj = (const float*)d_in[3];
  const float* bproj = (const float*)d_in[4];
  float* out = (float*)d_out;

  bf16_t* WtQ = (bf16_t*)((char*)d_ws + WS_WTQ);
  bf16_t* WtP = (bf16_t*)((char*)d_ws + WS_WTP);

  // 1) transpose+convert weights to bf16 (65536 elements total)
  {
    dim3 grid((C3 * CDIM + CDIM * CDIM + 255) / 256);
    dim3 block(256);
    hipLaunchKernelGGL(wprep_kernel, grid, block, 0, stream,
                       Wqkv, Wproj, WtQ, WtP);
  }
  // 2) fused attention: one workgroup per (b, w) column
  {
    dim3 grid(BATCH * WDIM);
    dim3 block(256);  // 8 wave32 waves
    hipLaunchKernelGGL(axial_attn_kernel, grid, block, SMEM_BYTES, stream,
                       x, WtQ, bqkv, WtP, bproj, out);
  }
}
